// Euclidean_20658792694341
// MI455X (gfx1250) — matrix operand order repaired and checked by reading, not compiled, not run on hardware
//
#include <hip/hip_runtime.h>

// ---------------------------------------------------------------------------
// Euclidean distance matrix (cdist p=2) via f16 WMMA GEMM on gfx1250.
//   d2 = ||a||^2 + ||b||^2 - 2 a.b ;  out = sqrt(max(d2, 0))
// Norms computed exactly in f32; cross term via v_wmma_f32_16x16x32_f16.
// Register-double-buffered k-chunks: next chunk's global loads are issued
// before the current chunk's WMMA phase, so load latency overlaps compute.
// ---------------------------------------------------------------------------

typedef _Float16 v4h  __attribute__((ext_vector_type(4)));
typedef _Float16 v8h  __attribute__((ext_vector_type(8)));
typedef _Float16 v16h __attribute__((ext_vector_type(16)));
typedef float    v8f  __attribute__((ext_vector_type(8)));

#define K_DIM 256   // feature dimension (from reference)
#define BT    128   // block tile edge (M and N)
#define KC    64    // K-chunk resident in LDS
#define LDH   72    // LDS row stride in halves (64 + 8 pad -> bank-conflict-free)

// ---------------------------------------------------------------------------
// Kernel 1: exact f32 squared row norms of x1 (rows 0..N-1) and x2 (N..N+M-1).
// One wave per row: 8 floats/lane, then a 5-step shfl_xor tree (wave32).
// ---------------------------------------------------------------------------
__global__ __launch_bounds__(256)
void cdist_rownorm_kernel(const float* __restrict__ x1,
                          const float* __restrict__ x2,
                          float* __restrict__ sq, int nRows) {
  const int lane = threadIdx.x & 31;
  const int wave = threadIdx.x >> 5;
  const int row  = blockIdx.x * 8 + wave;
  const float* src = (row < nRows) ? (x1 + (size_t)row * K_DIM)
                                   : (x2 + (size_t)(row - nRows) * K_DIM);
  float s = 0.f;
#pragma unroll
  for (int j = 0; j < 2; ++j) {
    const float4 v = *(const float4*)(src + lane * 4 + j * 128);
    s += v.x * v.x + v.y * v.y + v.z * v.z + v.w * v.w;
  }
#pragma unroll
  for (int off = 16; off > 0; off >>= 1) s += __shfl_xor(s, off, 32);
  if (lane == 0) sq[row] = s;
}

// ---------------------------------------------------------------------------
// Kernel 2: 128x128 output tile per 256-thread block (8 waves).
// Wave grid 2x4: each wave computes 64 rows x 32 cols = 4x2 WMMA 16x16 tiles.
// K walked in 64-wide LDS-resident chunks; f32->f16 conversion in the fill.
// ---------------------------------------------------------------------------
__global__ __launch_bounds__(256)
void cdist_wmma_kernel(const float* __restrict__ x1,
                       const float* __restrict__ x2,
                       const float* __restrict__ sq1,
                       const float* __restrict__ sq2,
                       float* __restrict__ out, int M) {
  __shared__ _Float16 As[BT * LDH];   // 18 KB
  __shared__ _Float16 Bs[BT * LDH];   // 18 KB
  __shared__ float sqA[BT], sqB[BT];

  const int tid     = threadIdx.x;
  const int lane    = tid & 31;
  const int l16     = lane & 15;
  const int lhalf   = lane >> 4;            // 0 or 1 (half-wave)
  const int wave    = tid >> 5;
  const int waveRow = (wave >> 2) * 64;     // 2 wave-rows of 64
  const int waveCol = (wave & 3) * 32;      // 4 wave-cols of 32
  const int rowBase = blockIdx.y * BT;
  const int colBase = blockIdx.x * BT;

  if (tid < BT) sqA[tid]      = sq1[rowBase + tid];
  else          sqB[tid - BT] = sq2[colBase + (tid - BT)];

  v8f acc[4][2] = {};   // 64 accumulator VGPRs

  // Per-thread fill assignment: 2048 float4 per matrix chunk -> 8 per thread.
  // idx = tid + j*256 ; row = idx/16 ; float4-col = idx%16.
  float4 ra[8], rb[8];  // register staging (double buffer vs. LDS phase)

  auto load_chunk = [&](int kc) {
#pragma unroll
    for (int j = 0; j < 8; ++j) {
      const int idx = tid + j * 256;
      const int r   = idx >> 4;
      const int c4  = idx & 15;
      ra[j] = *(const float4*)(x1 + (size_t)(rowBase + r) * K_DIM + kc + c4 * 4);
      rb[j] = *(const float4*)(x2 + (size_t)(colBase + r) * K_DIM + kc + c4 * 4);
    }
  };

  load_chunk(0);  // prologue: chunk 0 in flight

  for (int kc = 0; kc < K_DIM; kc += KC) {
    __syncthreads();    // previous chunk's fragment reads done before refill

    // Drain staged registers into LDS (f32 -> f16 conversion happens here;
    // this is where the wait for the in-flight loads lands).
#pragma unroll
    for (int j = 0; j < 8; ++j) {
      const int idx = tid + j * 256;
      const int r   = idx >> 4;
      const int c4  = idx & 15;
      v4h ha = {(_Float16)ra[j].x, (_Float16)ra[j].y, (_Float16)ra[j].z, (_Float16)ra[j].w};
      v4h hb = {(_Float16)rb[j].x, (_Float16)rb[j].y, (_Float16)rb[j].z, (_Float16)rb[j].w};
      *(v4h*)&As[r * LDH + c4 * 4] = ha;
      *(v4h*)&Bs[r * LDH + c4 * 4] = hb;
    }
    __syncthreads();

    // Kick off the next chunk's global loads; they overlap the WMMA phase.
    if (kc + KC < K_DIM) load_chunk(kc + KC);

#pragma unroll
    for (int kk = 0; kk < KC; kk += 32) {
      // B fragments: lane holds X2 row (n0 + lane%16), K-interleave per ISA:
      // halves 0-7 at kk + lhalf*8, halves 8-15 at kk + 16 + lhalf*8.
      v16h bf[2];
#pragma unroll
      for (int nt = 0; nt < 2; ++nt) {
        const _Float16* p = &Bs[(waveCol + nt * 16 + l16) * LDH + kk + (lhalf << 3)];
        v8h lo = *(const v8h*)p;
        v8h hi = *(const v8h*)(p + 16);
        bf[nt] = __builtin_shufflevector(lo, hi, 0, 1, 2, 3, 4, 5, 6, 7,
                                                 8, 9, 10, 11, 12, 13, 14, 15);
      }
      // A fragments: lane holds X1 row (m0 + lane%16), K-interleave per ISA:
      // halves 0-7 at kk + lhalf*8, halves 8-15 at kk + 16 + lhalf*8.
#pragma unroll
      for (int mt = 0; mt < 4; ++mt) {
        const _Float16* p = &As[(waveRow + mt * 16 + l16) * LDH + kk + (lhalf << 3)];
        v8h lo = *(const v8h*)p;
        v8h hi = *(const v8h*)(p + 16);
        v16h af = __builtin_shufflevector(lo, hi, 0, 1, 2, 3, 4, 5, 6, 7,
                                                  8, 9, 10, 11, 12, 13, 14, 15);
#pragma unroll
        for (int nt = 0; nt < 2; ++nt) {
          acc[mt][nt] = __builtin_amdgcn_wmma_f32_16x16x32_f16(
              false, af, false, bf[nt], (short)0, acc[mt][nt], false, false);
        }
      }
    }
  }

  // Epilogue: C/D layout -> element (vgpr v, lane): M = v + 8*(lane>=16), N = lane%16.
#pragma unroll
  for (int mt = 0; mt < 4; ++mt) {
#pragma unroll
    for (int nt = 0; nt < 2; ++nt) {
#pragma unroll
      for (int v = 0; v < 8; ++v) {
        const int r = waveRow + mt * 16 + v + (lhalf << 3);
        const int c = waveCol + nt * 16 + l16;
        float d2 = fmaf(-2.0f, acc[mt][nt][v], sqA[r] + sqB[c]);
        d2 = fmaxf(d2, 0.0f);
        out[(size_t)(rowBase + r) * (size_t)M + (size_t)(colBase + c)] = sqrtf(d2);
      }
    }
  }
}

// ---------------------------------------------------------------------------
extern "C" void kernel_launch(void* const* d_in, const int* in_sizes, int n_in,
                              void* d_out, int out_size, void* d_ws, size_t ws_size,
                              hipStream_t stream) {
  const float* x1 = (const float*)d_in[0];
  const float* x2 = (const float*)d_in[1];
  float* out = (float*)d_out;
  float* sq  = (float*)d_ws;             // [N + M] f32 squared norms

  const int N = in_sizes[0] / K_DIM;     // 8192
  const int M = in_sizes[1] / K_DIM;     // 8192

  // Pass 1: exact f32 row norms (8 rows per 256-thread block).
  cdist_rownorm_kernel<<<dim3((unsigned)((N + M) / 8)), dim3(256), 0, stream>>>(
      x1, x2, sq, N);

  // Pass 2: tiled WMMA distance matrix.
  dim3 grid((unsigned)(M / BT), (unsigned)(N / BT));
  cdist_wmma_kernel<<<grid, dim3(256), 0, stream>>>(x1, x2, sq, sq + N, out, M);
}
